// SigmaFlow_21268678050305
// MI455X (gfx1250) — matrix operand order repaired and checked by reading, not compiled, not run on hardware
//
#include <hip/hip_runtime.h>
#include <math.h>

// ---------------------------------------------------------------------------
// SigmaFlow on MI455X (gfx1250, wave32).
// 4 fused launches (one per NL iteration): conv3x3(9ch->16ch) and the 16->16
// per-pixel MLP run on v_wmma_f32_16x16x32_f16. LayerNorm + 16->3 head are
// folded into 5 per-pixel dot products (1 shuffle each); anisotropic stencil
// reads the LDS v-tile.
//
// Implicit-GEMM K-ordering matches the LDS layout so each lane's A-fragment
// slots are contiguous float2 LDS loads (no branches):
//   LDS tile: 10 ch/pixel (8 state, ch8 = padval, ch9 dummy), y-major.
//   Fragment f = conv row kh=f; slot s = kw*10 + cin (30 live slots, B=0 on
//   s=9,19,29..31 makes those A values don't-cares).
// ---------------------------------------------------------------------------

typedef __attribute__((ext_vector_type(16))) _Float16 v16h;
typedef __attribute__((ext_vector_type(8)))  float    v8f;

constexpr int Wd  = 1024;
constexpr int Hd  = 1024;
constexpr int C1  = 8;          // state channels
constexpr int CP  = 10;         // LDS channels/pixel: 8 state + pad + dummy
constexpr int C2  = 16;         // conv out / MLP width
constexpr int TLX = 16;         // tile x
constexpr int TLY = 32;         // tile y
constexpr int NTHREADS = 256;   // 8 wave32
constexpr int LVN = (TLX + 2) * (TLY + 2) * CP + 8;  // +8: harmless overrun pad

// Convert 8 contiguous floats (8B-aligned) into fragment elements [e0, e0+8).
__device__ __forceinline__ void cvt8(v16h& frag, int e0, const float* src) {
  const float2* p = reinterpret_cast<const float2*>(src);
  #pragma unroll
  for (int i = 0; i < 4; ++i) {
    float2 t = p[i];
    frag[e0 + 2 * i]     = (_Float16)t.x;
    frag[e0 + 2 * i + 1] = (_Float16)t.y;
  }
}

__global__ __launch_bounds__(NTHREADS)
void sigma_iter_kernel(const float* __restrict__ vin,
                       float* __restrict__ vout,
                       const float* __restrict__ conv_w,  // [16][9][3][3]
                       const float* __restrict__ conv_b,  // [16]
                       const float* __restrict__ w1,      // [16][16]
                       const float* __restrict__ b1,      // [16]
                       const float* __restrict__ ln_g,    // [16]
                       const float* __restrict__ ln_b,    // [16]
                       const float* __restrict__ w2,      // [3][16]
                       const float* __restrict__ b2,      // [3]
                       const float* __restrict__ mass,    // scalar
                       float padval)                      // = iteration index
{
  __shared__ float lv[LVN];                 // v tile + halo, 10 ch  (~24.5 KB)
  __shared__ float met[TLX * TLY * 4];      // a,c,b,rdet             (8 KB)
  __shared__ float ysc[8][16 * 16];         // per-wave transpose     (8 KB)

  const int tid  = threadIdx.x;
  const int wave = tid >> 5;
  const int lane = tid & 31;
  const int hi   = lane >> 4;
  const int nn   = lane & 15;     // C/D col (channel) == A row (pixel)
  const int bx0  = blockIdx.x * TLX;
  const int by0  = blockIdx.y * TLY;

  #define LV(x, y, ch) lv[(((x) * (TLY + 2)) + (y)) * CP + (ch)]

  // ---- Phase 0: cooperative tile + replicate-halo load -------------------
  for (int idx = tid; idx < (TLX + 2) * (TLY + 2); idx += NTHREADS) {
    int hx = idx / (TLY + 2), hy = idx % (TLY + 2);
    int gx = bx0 + hx - 1;  gx = gx < 0 ? 0 : (gx >= Wd ? Wd - 1 : gx);
    int gy = by0 + hy - 1;  gy = gy < 0 ? 0 : (gy >= Hd ? Hd - 1 : gy);
    const float4* src = reinterpret_cast<const float4*>(vin + (size_t)(gx * Hd + gy) * C1);
    float4 a = src[0];
    float4 b = src[1];
    float2* d2 = reinterpret_cast<float2*>(&LV(hx, hy, 0));  // 8B aligned
    d2[0] = make_float2(a.x, a.y);
    d2[1] = make_float2(a.z, a.w);
    d2[2] = make_float2(b.x, b.y);
    d2[3] = make_float2(b.z, b.w);
    d2[4] = make_float2(padval, 0.0f);      // ch8 = pad channel, ch9 dummy
  }
  __syncthreads();

  // ---- Weight B-fragments (once per block) -------------------------------
  // Conv fragment f (kh=f): B[s][n] = conv_w[n][cin][f][kw], s = kw*10+cin;
// hw slot for element e is s = 8*hi + e (e<8), 16 + 8*hi + (e-8) (e>=8).
  v16h bw[3];
  #pragma unroll
  for (int f = 0; f < 3; ++f) {
    #pragma unroll
    for (int e = 0; e < 16; ++e) {
      int s = (e < 8) ? (8 * hi + e) : (16 + 8 * hi + (e - 8));
      int kw = s / CP, cin = s % CP;
      float wv = (kw < 3 && cin < 9) ? conv_w[((nn * 9 + cin) * 3 + f) * 3 + kw] : 0.0f;
      bw[f][e] = (_Float16)wv;
    }
  }
  // MLP1: B[k][n] = w1[n][k], K 16..31 zero (elements e>=8 hold K>=16).
  v16h bw1;
  #pragma unroll
  for (int e = 0; e < 16; ++e)
    bw1[e] = (e >= 8) ? (_Float16)0.0f : (_Float16)w1[nn * C2 + 8 * hi + e];

  const float cb  = conv_b[nn];
  const float bb1 = b1[nn];

  // LN+head folding: m_d = inv*(dot(h,G_d) - mu*S_d) + T_d,
  //   G_d = ln_g .* w2_d,  S_d = sum(G_d),  T_d = dot(ln_b, w2_d) + b2_d.
  float G0[8], G1[8], G2[8];
  #pragma unroll
  for (int i = 0; i < 8; ++i) {
    int c = 8 * hi + i;           // this lane's channel half
    float g = ln_g[c];
    G0[i] = g * w2[0 * C2 + c];
    G1[i] = g * w2[1 * C2 + c];
    G2[i] = g * w2[2 * C2 + c];
  }
  float S0 = 0.f, S1 = 0.f, S2 = 0.f;
  float T0 = b2[0], T1 = b2[1], T2 = b2[2];
  #pragma unroll
  for (int c = 0; c < C2; ++c) {
    float g = ln_g[c], lbv = ln_b[c];
    S0 += g * w2[0 * C2 + c];  T0 += lbv * w2[0 * C2 + c];
    S1 += g * w2[1 * C2 + c];  T1 += lbv * w2[1 * C2 + c];
    S2 += g * w2[2 * C2 + c];  T2 += lbv * w2[2 * C2 + c];
  }

  // ---- Phase 1: 16-pixel groups; wave w owns groups [4w, 4w+4) -----------
  for (int gi = 0; gi < 4; ++gi) {
    const int g   = wave * 4 + gi;
    const int ix  = g >> 1;            // tile x of group
    const int iy0 = (g & 1) * 16;
    const int py  = iy0 + nn;          // this lane's pixel (A row M)

    // A fragments: fragment f = conv row kh=f. Contiguous 30-float run in
    // LDS starting at (ix+f, py). Lane holds runs s=[8hi,+8) and [16+8hi,+8).
    v16h af[3];
    #pragma unroll
    for (int f = 0; f < 3; ++f) {
      const float* row = &LV(ix + f, py, 0);
      cvt8(af[f], 0, row + 8 * hi);
      cvt8(af[f], 8, row + 16 + 8 * hi);
    }
    v8f acc = {};
    #pragma unroll
    for (int f = 0; f < 3; ++f)
      acc = __builtin_amdgcn_wmma_f32_16x16x32_f16(
          false, af[f], false, bw[f], (short)0, acc, false, false);

    // Transpose y through wave-private LDS (C/D row M = j + 8*hi, col nn).
    float* ys = &ysc[wave][0];
    #pragma unroll
    for (int j = 0; j < 8; ++j)
      ys[(j + 8 * hi) * 16 + nn] = acc[j] + cb;
    __builtin_amdgcn_wave_barrier();   // LDS is in-order within a wave

    v16h ha;
    cvt8(ha, 0, &ys[nn * 16 + 8 * hi]);   // K = 8hi .. 8hi+7
    #pragma unroll
    for (int e = 8; e < 16; ++e) ha[e] = (_Float16)0.0f;  // K>=16 dead
    v8f hv = {};
    hv = __builtin_amdgcn_wmma_f32_16x16x32_f16(
        false, ha, false, bw1, (short)0, hv, false, false);

    // bias + ReLU, back through the transpose scratch (overwrite is safe:
    // LDS in-order per wave; barrier stops compiler reordering).
    __builtin_amdgcn_wave_barrier();
    #pragma unroll
    for (int j = 0; j < 8; ++j) {
      float h = hv[j] + bb1;
      ys[(j + 8 * hi) * 16 + nn] = h > 0.0f ? h : 0.0f;
    }
    __builtin_amdgcn_wave_barrier();

    // Per-pixel stats: lane (nn,hi) owns pixel nn, channels [8hi, 8hi+8).
    const float* hrow = &ys[nn * 16 + 8 * hi];
    float s = 0.f, s2 = 0.f, d0 = 0.f, d1 = 0.f, d2 = 0.f;
    #pragma unroll
    for (int i = 0; i < 8; ++i) {
      float h = hrow[i];
      s  += h;
      s2 += h * h;
      d0 += h * G0[i];
      d1 += h * G1[i];
      d2 += h * G2[i];
    }
    s  += __shfl_xor(s,  16, 32);      // combine channel halves
    s2 += __shfl_xor(s2, 16, 32);
    d0 += __shfl_xor(d0, 16, 32);
    d1 += __shfl_xor(d1, 16, 32);
    d2 += __shfl_xor(d2, 16, 32);

    float mu  = s * 0.0625f;
    float var = s2 * 0.0625f - mu * mu;
    float inv = rsqrtf(var + 1e-5f);
    float m0 = inv * (d0 - mu * S0) + T0;
    float m1 = inv * (d1 - mu * S1) + T1;
    float m2 = inv * (d2 - mu * S2) + T2;

    float scale = 1.0f - tanhf(fabsf(m1)) * 0.9f;   // root_det
    float l1    = 1.0f - tanhf(fabsf(m0)) * 0.9f;
    float Delta = (1.0f - l1 * l1) / l1;
    float alpha = tanhf(m2) * 1.57079632679f;
    float cs  = cosf(alpha);
    float c2v = cs * cs;
    float sc  = sinf(alpha) * cs;
    float ma  = l1 + (1.0f - c2v) * Delta;
    float mc  = l1 + c2v * Delta;
    float mb  = Delta * sc;

    if (hi == 0) {                     // one writer per pixel
      int p = ix * TLY + iy0 + nn;
      float4* mp = reinterpret_cast<float4*>(&met[p * 4]);
      *mp = make_float4(ma, mc, mb, scale);
    }
  }
  __syncthreads();

  // ---- Phase 2: anisotropic stencil + state update -----------------------
  const float one_mass = 1.0f + mass[0];
  for (int p = tid; p < TLX * TLY; p += NTHREADS) {
    int ix = p / TLY, iy = p % TLY;
    float ma   = met[p * 4 + 0];
    float mc   = met[p * 4 + 1];
    float mb2  = 2.0f * met[p * 4 + 2];
    float coef = 0.5f / met[p * 4 + 3];
    float* outp = vout + (size_t)((bx0 + ix) * Hd + (by0 + iy)) * C1;
    #pragma unroll
    for (int ch = 0; ch < C1; ++ch) {
      float vc  = LV(ix + 1, iy + 1, ch);
      float vxx = LV(ix + 2, iy + 1, ch) + LV(ix,     iy + 1, ch) - 2.0f * vc;
      float vyy = LV(ix + 1, iy + 2, ch) + LV(ix + 1, iy,     ch) - 2.0f * vc;
      float vxy = 0.25f * (LV(ix + 2, iy + 2, ch) + LV(ix,     iy,     ch)
                         - LV(ix + 2, iy,     ch) - LV(ix,     iy + 2, ch));
      outp[ch] = one_mass * vc + coef * (ma * vxx + mc * vyy + mb2 * vxy);
    }
  }
  #undef LV
}

extern "C" void kernel_launch(void* const* d_in, const int* in_sizes, int n_in,
                              void* d_out, int out_size, void* d_ws, size_t ws_size,
                              hipStream_t stream) {
  const float* x      = (const float*)d_in[0];
  const float* conv_w = (const float*)d_in[1];
  const float* conv_b = (const float*)d_in[2];
  const float* w1     = (const float*)d_in[3];
  const float* b1     = (const float*)d_in[4];
  const float* ln_g   = (const float*)d_in[5];
  const float* ln_b   = (const float*)d_in[6];
  const float* w2     = (const float*)d_in[7];
  const float* b2     = (const float*)d_in[8];
  const float* mass   = (const float*)d_in[9];

  float* bufA = (float*)d_ws;      // 32 MB ping buffer
  float* out  = (float*)d_out;

  const float* src[4] = { x, bufA, out, bufA };
  float*       dst[4] = { bufA, out, bufA, out };

  dim3 grid(Wd / TLX, Hd / TLY);
  dim3 block(NTHREADS);
  for (int i = 0; i < 4; ++i) {
    sigma_iter_kernel<<<grid, block, 0, stream>>>(
        src[i], dst[i], conv_w, conv_b, w1, b1, ln_g, ln_b, w2, b2, mass,
        (float)i);
  }
}
